// Net_ddi_15307263442979
// MI455X (gfx1250) — matrix operand order repaired and checked by reading, not compiled, not run on hardware
//
#include <hip/hip_runtime.h>

#define NN 768
#define EE 24576
#define PP 65536
#define DD 64

typedef __attribute__((ext_vector_type(16))) _Float16 v16h;
typedef __attribute__((ext_vector_type(8)))  _Float16 v8h;
typedef __attribute__((ext_vector_type(8)))  float    v8f;

static __device__ inline v16h cat8(v8h lo, v8h hi) {
    return __builtin_shufflevector(lo, hi, 0, 1, 2, 3, 4, 5, 6, 7,
                                   8, 9, 10, 11, 12, 13, 14, 15);
}

// ---------------------------------------------------------------------------
// Graph setup
// ---------------------------------------------------------------------------
__global__ void k_zero(float* p, int n) {
    int i = blockIdx.x * blockDim.x + threadIdx.x;
    if (i < n) p[i] = 0.f;
}

__global__ void k_graph(const int* __restrict__ src, const int* __restrict__ dst,
                        float* __restrict__ eim, float* __restrict__ deg) {
    int e = blockIdx.x * blockDim.x + threadIdx.x;
    if (e < EE) {
        int s = src[e], d = dst[e];
        eim[s * NN + d] = 1.0f;
        atomicAdd(&deg[d], 1.0f);
    }
}

__global__ void k_dinv(const float* __restrict__ deg, float* __restrict__ dinv) {
    int i = blockIdx.x * blockDim.x + threadIdx.x;
    if (i < NN) dinv[i] = rsqrtf(deg[i] + 1.0f);  // +1 self loop
}

__global__ void k_gather_emb(const int* __restrict__ ids, const float* __restrict__ emb,
                             float* __restrict__ x) {
    int t = blockIdx.x * blockDim.x + threadIdx.x;  // NN*DD
    int i = t >> 6, c = t & 63;
    x[t] = emb[ids[i] * DD + c];
}

// ---------------------------------------------------------------------------
// GCN layer pieces (tiny: stay on VALU)
// ---------------------------------------------------------------------------
__global__ void k_gemm64(const float* __restrict__ x, const float* __restrict__ w,
                         float* __restrict__ xw) {
    int t = blockIdx.x * blockDim.x + threadIdx.x;  // NN*DD
    int i = t >> 6, d = t & 63;
    float acc = 0.f;
#pragma unroll
    for (int c = 0; c < DD; c++) acc += x[i * DD + c] * w[c * DD + d];
    xw[t] = acc;
}

__global__ void k_selfinit(const float* __restrict__ xw, const float* __restrict__ dinv,
                           const float* __restrict__ b, float* __restrict__ agg) {
    int t = blockIdx.x * blockDim.x + threadIdx.x;  // NN*DD
    int i = t >> 6, d = t & 63;
    float di = dinv[i];
    agg[t] = xw[t] * di * di + b[d];
}

__global__ void k_edges(const int* __restrict__ src, const int* __restrict__ dst,
                        const float* __restrict__ xw, const float* __restrict__ dinv,
                        float* __restrict__ agg) {
    int t = blockIdx.x * blockDim.x + threadIdx.x;  // EE*16
    int e = t >> 4, q = t & 15;
    int s = src[e], d = dst[e];
    float cf = dinv[s] * dinv[d];
    int c0 = q * 4;
#pragma unroll
    for (int k = 0; k < 4; k++)
        atomicAdd(&agg[d * DD + c0 + k], xw[s * DD + c0 + k] * cf);
}

// GraphNorm over node axis + ReLU; one block per channel d
__global__ void k_gnorm_nodes(const float* __restrict__ a, const float* __restrict__ w,
                              const float* __restrict__ b, const float* __restrict__ ms,
                              float* __restrict__ out) {
    int d = blockIdx.x, tid = threadIdx.x;
    __shared__ float l1[256], l2[256];
    __shared__ float ssc, ssh;
    float s1 = 0.f, s2 = 0.f;
    for (int i = tid; i < NN; i += 256) {
        float v = a[i * DD + d];
        s1 += v; s2 += v * v;
    }
    l1[tid] = s1; l2[tid] = s2; __syncthreads();
    for (int st = 128; st > 0; st >>= 1) {
        if (tid < st) { l1[tid] += l1[tid + st]; l2[tid] += l2[tid + st]; }
        __syncthreads();
    }
    if (tid == 0) {
        float mean = l1[0] * (1.0f / NN);
        float m = ms[d];
        float var = l2[0] * (1.0f / NN) - mean * mean * m * (2.f - m);
        float r = rsqrtf(var + 1e-5f);
        ssc = w[d] * r;
        ssh = b[d] - mean * m * ssc;
    }
    __syncthreads();
    float sc = ssc, sh = ssh;
    for (int i = tid; i < NN; i += 256) {
        float y = a[i * DD + d] * sc + sh;
        out[i * DD + d] = fmaxf(y, 0.f);
    }
}

// ---------------------------------------------------------------------------
// Precision-prep kernels: f16 copy of x, f16-transposed weights
// ---------------------------------------------------------------------------
__global__ void k_cvt_xh(const float* __restrict__ x, _Float16* __restrict__ xh) {
    int t = blockIdx.x * blockDim.x + threadIdx.x;  // NN*DD
    xh[t] = (_Float16)x[t];
}

// wt[d][c] = w[c][d] for c < 64 (m1/m2 first 64 rows)
__global__ void k_prep_wT64(const float* __restrict__ w, _Float16* __restrict__ wt) {
    int t = blockIdx.x * blockDim.x + threadIdx.x;  // 4096
    int d = t >> 6, c = t & 63;
    wt[d * DD + c] = (_Float16)w[c * DD + d];
}

// w3t[d][q], q<64 -> m3 row q (outer-product channels), q>=64 -> m3 row q+1 (prod)
__global__ void k_prep_w3t(const float* __restrict__ w, _Float16* __restrict__ wt) {
    int t = blockIdx.x * blockDim.x + threadIdx.x;  // 8192
    int d = t >> 7, q = t & 127;
    int row = q < 64 ? q : q + 1;
    wt[d * 128 + q] = (_Float16)w[row * DD + d];
}

// ---------------------------------------------------------------------------
// WMMA pair-MLP (m1 or m2): rows r against fixed node f.
//   A[r,c] = xh[f,c]*xh[r,c] (K=64, 2 WMMA steps), eim channel as epilogue.
//   out[((d*768)+f)*768 + r]  (f16, packed 16B stores per lane)
//   EIM_MODE 0: e = eim[f*768+r]   (x1: f=i, r=k)
//   EIM_MODE 1: e = eim[r*768+f]   (x2: f=j, r=k)
// ---------------------------------------------------------------------------
template <int EIM_MODE>
__global__ __launch_bounds__(256) void k_mlp_wmma(const _Float16* __restrict__ xh,
                                                  const float* __restrict__ eim,
                                                  const _Float16* __restrict__ wt,
                                                  const float* __restrict__ wfull,
                                                  const float* __restrict__ bias,
                                                  _Float16* __restrict__ out) {
    int tid = threadIdx.x;
    int lane = tid & 31;
    int w = tid >> 5;
    int f = blockIdx.y;
    int rt = blockIdx.x * 128 + w * 16;
    int lrow = lane & 15;
    int hi = lane >> 4;
    int koffA = hi ? 8 : 0;
    int koffB = hi ? 16 : 0;
    int r = rt + lrow;

    v8f acc[4];
#pragma unroll
    for (int tn = 0; tn < 4; tn++) {
        float bv = bias[tn * 16 + lrow];
#pragma unroll
        for (int q = 0; q < 8; q++) acc[tn][q] = bv;
    }
    const _Float16* xf = xh + f * DD;
    const _Float16* xr = xh + r * DD;
#pragma unroll
    for (int s = 0; s < 2; s++) {
        int k0 = s * 32;
        v8h flo = *(const v8h*)(xf + k0 + koffA);
        v8h fhi = *(const v8h*)(xf + k0 + koffA + 16);
        v8h rlo = *(const v8h*)(xr + k0 + koffA);
        v8h rhi = *(const v8h*)(xr + k0 + koffA + 16);
        v16h a = cat8(flo * rlo, fhi * rhi);
#pragma unroll
        for (int tn = 0; tn < 4; tn++) {
            const _Float16* pb = wt + (tn * 16 + lrow) * DD + k0 + koffB;
            v16h b = cat8(*(const v8h*)(pb - koffB + koffA), *(const v8h*)(pb - koffB + koffA + 16));
            acc[tn] = __builtin_amdgcn_wmma_f32_16x16x32_f16(
                false, a, false, b, (short)0, acc[tn], false, false);
        }
    }
    // eim channel (row 64 of original weights), ReLU, packed f16 store
    float ev[8];
#pragma unroll
    for (int v = 0; v < 8; v++) {
        int rr = rt + v + hi * 8;
        ev[v] = (EIM_MODE == 1) ? eim[rr * NN + f] : eim[f * NN + rr];
    }
#pragma unroll
    for (int tn = 0; tn < 4; tn++) {
        int d = tn * 16 + lrow;
        float wv = wfull[64 * DD + d];
        v8h y;
#pragma unroll
        for (int v = 0; v < 8; v++) {
            float t = acc[tn][v] + ev[v] * wv;
            y[v] = (_Float16)fmaxf(t, 0.f);
        }
        _Float16* po = out + ((size_t)d * NN + f) * NN + rt + hi * 8;
        *(v8h*)po = y;
    }
}

// ---------------------------------------------------------------------------
// WMMA einsum: prod[c][i][j] (f16) = x1s[c][i][:] @ x2s[c][j][:], K=768
// Each wave: one 64x64 output tile (16 accumulator fragments).
// ---------------------------------------------------------------------------
__global__ __launch_bounds__(256) void k_einsum(const _Float16* __restrict__ x1s,
                                                const _Float16* __restrict__ x2s,
                                                _Float16* __restrict__ prodh) {
    int wave = (blockIdx.x * blockDim.x + threadIdx.x) >> 5;
    int lane = threadIdx.x & 31;
    int c = wave / 144;
    int t = wave % 144;                    // 12x12 tiles of 64x64
    int i0 = (t / 12) * 64, j0 = (t % 12) * 64;
    size_t cbase = (size_t)c * NN * NN;
    int lrow  = lane & 15;
    int hi    = lane >> 4;
    int koffA = hi ? 8 : 0;
    int koffB = hi ? 16 : 0;

    v8f zero;
#pragma unroll
    for (int q = 0; q < 8; q++) zero[q] = 0.f;
    v8f acc[4][4];
#pragma unroll
    for (int ti = 0; ti < 4; ti++)
#pragma unroll
        for (int tj = 0; tj < 4; tj++) acc[ti][tj] = zero;

    const _Float16* pA = x1s + cbase;
    const _Float16* pB = x2s + cbase;
    for (int k0 = 0; k0 < NN; k0 += 32) {
        v16h a[4], b[4];
#pragma unroll
        for (int ti = 0; ti < 4; ti++) {
            const _Float16* p = pA + (size_t)(i0 + ti * 16 + lrow) * NN + k0 + koffA;
            v8h lo = *(const v8h*)p;
            v8h hh = *(const v8h*)(p + 16);
            a[ti] = cat8(lo, hh);
        }
#pragma unroll
        for (int tj = 0; tj < 4; tj++) {
            const _Float16* p = pB + (size_t)(j0 + tj * 16 + lrow) * NN + k0 + koffB;
            b[tj] = cat8(*(const v8h*)(p - koffB + koffA), *(const v8h*)(p - koffB + koffA + 16));
        }
#pragma unroll
        for (int ti = 0; ti < 4; ti++)
#pragma unroll
            for (int tj = 0; tj < 4; tj++)
                acc[ti][tj] = __builtin_amdgcn_wmma_f32_16x16x32_f16(
                    false, a[ti], false, b[tj], (short)0, acc[ti][tj], false, false);
    }
    int rofs = hi ? 8 : 0;
#pragma unroll
    for (int ti = 0; ti < 4; ti++)
#pragma unroll
        for (int tj = 0; tj < 4; tj++) {
            int j = j0 + tj * 16 + lrow;
#pragma unroll
            for (int v = 0; v < 8; v++) {
                int i = i0 + ti * 16 + v + rofs;
                prodh[cbase + (size_t)i * NN + j] = (_Float16)acc[ti][tj][v];
            }
        }
}

// ---------------------------------------------------------------------------
// WMMA stage D: t[i][j][d] = concat(x_i*x_j, eim, prod) @ m3 + b3 (pre-norm)
// Fixed i per block; 128 j rows per block (8 waves x 16); K=128 via 4 WMMA
// steps (2 outer-product + 2 prod-from-LDS); eim channel as epilogue.
// ---------------------------------------------------------------------------
__global__ __launch_bounds__(256) void k_stageD_wmma(const _Float16* __restrict__ xh,
                                                     const float* __restrict__ eim,
                                                     const _Float16* __restrict__ prodh,
                                                     const _Float16* __restrict__ w3t,
                                                     const float* __restrict__ m3w,
                                                     const float* __restrict__ b3,
                                                     float* __restrict__ tbuf) {
    __shared__ _Float16 sP[128 * 72];      // rows j_local (stride 72 halves, 16B aligned)
    int tid = threadIdx.x;
    int i = blockIdx.y;
    int j0 = blockIdx.x * 128;
    // stage prod tile [j_local 0..127][c 0..63] into LDS (transpose from [c][i][j])
    for (int idx = tid; idx < 64 * 64; idx += 256) {
        int c = idx >> 6;
        int jl = (idx & 63) << 1;
        const _Float16* ps = prodh + ((size_t)c * NN + i) * NN + j0 + jl;
        _Float16 h0 = ps[0];
        _Float16 h1 = ps[1];
        sP[jl * 72 + c] = h0;
        sP[(jl + 1) * 72 + c] = h1;
    }
    __syncthreads();

    int lane = tid & 31;
    int w = tid >> 5;
    int jt = j0 + w * 16;
    int lrow = lane & 15;
    int hi = lane >> 4;
    int koffA = hi ? 8 : 0;
    int koffB = hi ? 16 : 0;
    int j = jt + lrow;

    v8f acc[4];
#pragma unroll
    for (int tn = 0; tn < 4; tn++) {
        float bv = b3[tn * 16 + lrow];
#pragma unroll
        for (int q = 0; q < 8; q++) acc[tn][q] = bv;
    }
    const _Float16* xi = xh + i * DD;
    const _Float16* xj = xh + j * DD;
    const _Float16* sprow = sP + (w * 16 + lrow) * 72;
#pragma unroll
    for (int s = 0; s < 4; s++) {
        int k0 = (s & 1) * 32;
        v16h a;
        if (s < 2) {
            v8h flo = *(const v8h*)(xi + k0 + koffA);
            v8h fhi = *(const v8h*)(xi + k0 + koffA + 16);
            v8h rlo = *(const v8h*)(xj + k0 + koffA);
            v8h rhi = *(const v8h*)(xj + k0 + koffA + 16);
            a = cat8(flo * rlo, fhi * rhi);
        } else {
            v8h lo = *(const v8h*)(sprow + k0 + koffA);
            v8h hh = *(const v8h*)(sprow + k0 + koffA + 16);
            a = cat8(lo, hh);
        }
#pragma unroll
        for (int tn = 0; tn < 4; tn++) {
            const _Float16* pb = w3t + (tn * 16 + lrow) * 128 + s * 32 + koffB;
            v16h b = cat8(*(const v8h*)(pb - koffB + koffA), *(const v8h*)(pb - koffB + koffA + 16));
            acc[tn] = __builtin_amdgcn_wmma_f32_16x16x32_f16(
                false, a, false, b, (short)0, acc[tn], false, false);
        }
    }
    // eim channel (m3 row 64) epilogue + f32 store of pre-norm t
    float ev[8];
#pragma unroll
    for (int v = 0; v < 8; v++) ev[v] = eim[i * NN + jt + v + hi * 8];
#pragma unroll
    for (int tn = 0; tn < 4; tn++) {
        int d = tn * 16 + lrow;
        float wv = m3w[64 * DD + d];
#pragma unroll
        for (int v = 0; v < 8; v++) {
            int jj = jt + v + hi * 8;
            tbuf[((size_t)i * NN + jj) * DD + d] = acc[tn][v] + ev[v] * wv;
        }
    }
}

// GraphNorm stats over axis 0 (i) for each (j,d) -> scale/shift
__global__ void k_reduce(const float* __restrict__ tbuf, const float* __restrict__ w,
                         const float* __restrict__ b, const float* __restrict__ ms,
                         float* __restrict__ scale, float* __restrict__ shift) {
    int j = blockIdx.x, tid = threadIdx.x;
    int d = tid & 63, g = tid >> 6;
    float s1 = 0.f, s2 = 0.f;
    for (int i = g * 192; i < (g + 1) * 192; i++) {
        float v = tbuf[((size_t)i * NN + j) * DD + d];
        s1 += v; s2 += v * v;
    }
    __shared__ float l1[256], l2[256];
    l1[tid] = s1; l2[tid] = s2; __syncthreads();
    if (tid < 64) {
        float t1 = l1[tid] + l1[tid + 64] + l1[tid + 128] + l1[tid + 192];
        float t2 = l2[tid] + l2[tid + 64] + l2[tid + 128] + l2[tid + 192];
        float mean = t1 * (1.0f / NN);
        float m = ms[d];
        float var = t2 * (1.0f / NN) - mean * mean * m * (2.f - m);
        float r = rsqrtf(var + 1e-5f);
        float sc = w[d] * r;
        scale[j * DD + d] = sc;
        shift[j * DD + d] = b[d] - mean * m * sc;
    }
}

// Final: norm+ReLU on the fly, symmetrize, gather pos pairs, apply lin
__global__ void k_gatherout(const int* __restrict__ pos, const float* __restrict__ tbuf,
                            const float* __restrict__ scale, const float* __restrict__ shift,
                            const float* __restrict__ lw, const float* __restrict__ lb,
                            float* __restrict__ out) {
    int p = blockIdx.x * blockDim.x + threadIdx.x;
    if (p >= PP) return;
    int i = pos[2 * p], j = pos[2 * p + 1];
    const float* t1 = tbuf + ((size_t)i * NN + j) * DD;
    const float* t2 = tbuf + ((size_t)j * NN + i) * DD;
    const float* scj = scale + j * DD; const float* shj = shift + j * DD;
    const float* sci = scale + i * DD; const float* shi = shift + i * DD;
    float acc = lb[0];
#pragma unroll 1
    for (int d = 0; d < DD; d++) {
        float y1 = fmaxf(t1[d] * scj[d] + shj[d], 0.f);
        float y2 = fmaxf(t2[d] * sci[d] + shi[d], 0.f);
        acc += (y1 + y2) * lw[d];
    }
    out[p] = acc;
}

// ---------------------------------------------------------------------------
extern "C" void kernel_launch(void* const* d_in, const int* in_sizes, int n_in,
                              void* d_out, int out_size, void* d_ws, size_t ws_size,
                              hipStream_t stream) {
    const int*   node_ids = (const int*)d_in[0];
    const int*   ei  = (const int*)d_in[1];
    const int*   src = ei;
    const int*   dst = ei + EE;
    const int*   pos = (const int*)d_in[2];
    const float* emb = (const float*)d_in[3];
    const float* gw0 = (const float*)d_in[4];  const float* gb0 = (const float*)d_in[5];
    const float* n0w = (const float*)d_in[6];  const float* n0b = (const float*)d_in[7];  const float* n0m = (const float*)d_in[8];
    const float* gw1 = (const float*)d_in[9];  const float* gb1 = (const float*)d_in[10];
    const float* n1w = (const float*)d_in[11]; const float* n1b = (const float*)d_in[12]; const float* n1m = (const float*)d_in[13];
    const float* m1w = (const float*)d_in[14]; const float* m1b = (const float*)d_in[15];
    const float* m2w = (const float*)d_in[16]; const float* m2b = (const float*)d_in[17];
    const float* m3w = (const float*)d_in[18]; const float* m3b = (const float*)d_in[19];
    const float* n3w = (const float*)d_in[20]; const float* n3b = (const float*)d_in[21]; const float* n3m = (const float*)d_in[22];
    const float* lw  = (const float*)d_in[23]; const float* lb  = (const float*)d_in[24];

    char* ws = (char*)d_ws;
    size_t off = 0;
    auto take = [&](size_t bytes) -> char* {
        char* p = ws + off;
        off = (off + bytes + 255) & ~(size_t)255;
        return p;
    };
    float*    f_x    = (float*)take((size_t)NN * DD * 4);
    float*    f_xw   = (float*)take((size_t)NN * DD * 4);
    float*    f_agg  = (float*)take((size_t)NN * DD * 4);
    float*    f_deg  = (float*)take((size_t)NN * 4);
    float*    f_dinv = (float*)take((size_t)NN * 4);
    float*    f_scal = (float*)take((size_t)NN * DD * 4);
    float*    f_shft = (float*)take((size_t)NN * DD * 4);
    float*    f_eim  = (float*)take((size_t)NN * NN * 4);
    _Float16* xh     = (_Float16*)take((size_t)NN * DD * 2);
    _Float16* w1t    = (_Float16*)take((size_t)DD * DD * 2);
    _Float16* w2t    = (_Float16*)take((size_t)DD * DD * 2);
    _Float16* w3t    = (_Float16*)take((size_t)DD * 128 * 2);
    _Float16* x1s    = (_Float16*)take((size_t)DD * NN * NN * 2);
    _Float16* x2s    = (_Float16*)take((size_t)DD * NN * NN * 2);
    _Float16* prodh  = (_Float16*)take((size_t)DD * NN * NN * 2);
    float*    tbuf   = (float*)take((size_t)NN * NN * DD * 4);

    // Graph structure
    k_zero<<<(NN * NN + 255) / 256, 256, 0, stream>>>(f_eim, NN * NN);
    k_zero<<<(NN + 255) / 256, 256, 0, stream>>>(f_deg, NN);
    k_graph<<<(EE + 255) / 256, 256, 0, stream>>>(src, dst, f_eim, f_deg);
    k_dinv<<<(NN + 255) / 256, 256, 0, stream>>>(f_deg, f_dinv);
    k_gather_emb<<<(NN * DD) / 256, 256, 0, stream>>>(node_ids, emb, f_x);

    // GCN layer 0
    k_gemm64<<<(NN * DD) / 256, 256, 0, stream>>>(f_x, gw0, f_xw);
    k_selfinit<<<(NN * DD) / 256, 256, 0, stream>>>(f_xw, f_dinv, gb0, f_agg);
    k_edges<<<(EE * 16) / 256, 256, 0, stream>>>(src, dst, f_xw, f_dinv, f_agg);
    k_gnorm_nodes<<<DD, 256, 0, stream>>>(f_agg, n0w, n0b, n0m, f_x);
    // GCN layer 1
    k_gemm64<<<(NN * DD) / 256, 256, 0, stream>>>(f_x, gw1, f_xw);
    k_selfinit<<<(NN * DD) / 256, 256, 0, stream>>>(f_xw, f_dinv, gb1, f_agg);
    k_edges<<<(EE * 16) / 256, 256, 0, stream>>>(src, dst, f_xw, f_dinv, f_agg);
    k_gnorm_nodes<<<DD, 256, 0, stream>>>(f_agg, n1w, n1b, n1m, f_x);

    // f16 prep (x copy + transposed weights)
    k_cvt_xh<<<(NN * DD) / 256, 256, 0, stream>>>(f_x, xh);
    k_prep_wT64<<<(DD * DD) / 256, 256, 0, stream>>>(m1w, w1t);
    k_prep_wT64<<<(DD * DD) / 256, 256, 0, stream>>>(m2w, w2t);
    k_prep_w3t<<<(DD * 128) / 256, 256, 0, stream>>>(m3w, w3t);

    // Pair MLPs on WMMA -> f16 x1s[c][i][k] / x2s[c][j][k]
    dim3 gm(NN / 128, NN);
    k_mlp_wmma<0><<<gm, 256, 0, stream>>>(xh, f_eim, w1t, m1w, m1b, x1s);
    k_mlp_wmma<1><<<gm, 256, 0, stream>>>(xh, f_eim, w2t, m2w, m2b, x2s);

    // Per-channel 768^3 GEMMs on WMMA: 64ch * 144 tiles = 9216 waves, 8/block
    k_einsum<<<(DD * 144) / 8, 256, 0, stream>>>(x1s, x2s, prodh);

    // m3 stage on WMMA (pre-norm t), norm stats, final gather
    dim3 gd(NN / 128, NN);
    k_stageD_wmma<<<gd, 256, 0, stream>>>(xh, f_eim, prodh, w3t, m3w, m3b, tbuf);
    k_reduce<<<NN, 256, 0, stream>>>(tbuf, n3w, n3b, n3m, f_scal, f_shft);
    k_gatherout<<<PP / 256, 256, 0, stream>>>(pos, tbuf, f_scal, f_shft, lw, lb, (float*)d_out);
}
